// CrossFeatureAttention_54520314856358
// MI455X (gfx1250) — compile-verified, matrix operand order checked by reading, not run on hardware
//
#include <hip/hip_runtime.h>

// ---------------------------------------------------------------------------
// CDNA5 (gfx1250) transformer-encoder forward.
//   L=4, B=8, T=1024, D=512, H=8, HD=64, FF=2048
// All GEMMs run on v_wmma_f32_16x16x32_bf16 (fp32 inputs converted to bf16;
// fp32 accumulate). GEMM stages A/W tiles into LDS with gfx1250 async
// global->LDS copies (ASYNCcnt + s_wait_asynccnt). Attention is fused per
// 16-query block with full-row softmax in LDS. Compute-bound workload
// (~275 GFLOP vs ~100MB HBM traffic at 23.3 TB/s): matrix-pipe bound.
// ---------------------------------------------------------------------------

typedef __bf16 bf16;
typedef __attribute__((ext_vector_type(16))) __bf16 v16bf;
typedef __attribute__((ext_vector_type(8)))  __bf16 v8bf;
typedef __attribute__((ext_vector_type(8)))  float  v8f;

union V16 { v16bf v; v8bf h[2]; };

#define B_  8
#define T_  1024
#define D_  512
#define H_  8
#define HD_ 64
#define FF_ 2048
#define L_  4
#define M_  (B_ * T_)          // 8192 token rows

// ---------------------------------------------------------------------------
// bf16 A/B fragment loader (16x32 tile, 16-bit element WMMA layout):
//   lanes 0-15  : row = lane,    K = {0..7, 16..23}
//   lanes 16-31 : row = lane-16, K = {8..15, 24..31}
// `row_base` points at element (row, 0) of a 32-wide row (LDS or global).
// ---------------------------------------------------------------------------
__device__ __forceinline__ v16bf make_frag(const bf16* row_base, int lane) {
  int fo = (lane >> 4) * 8;
  V16 u;
  u.h[0] = *(const v8bf*)(row_base + fo);
  u.h[1] = *(const v8bf*)(row_base + fo + 16);
  return u.v;
}

// ---------------------------------------------------------------------------
// gfx1250 async global->LDS 16-byte copy. Each lane copies 16B from its own
// global address into its own LDS offset (low 32 bits of a flat __shared__
// pointer == wave-relative LDS byte offset, per the flat-aperture rules).
// Tracked by ASYNCcnt; fence with s_wait_asynccnt before the barrier.
// ---------------------------------------------------------------------------
__device__ __forceinline__ void async_cp16(const bf16* g, const bf16* l) {
  unsigned lds          = (unsigned)(unsigned long long)(uintptr_t)l;
  unsigned long long ga = (unsigned long long)(uintptr_t)g;
  asm volatile("global_load_async_to_lds_b128 %0, %1, off"
               :: "v"(lds), "v"(ga)
               : "memory");
}
__device__ __forceinline__ void wait_async0() {
  asm volatile("s_wait_asynccnt 0" ::: "memory");
}

// ---------------------------------------------------------------------------
// f32 -> bf16 convert
// ---------------------------------------------------------------------------
__global__ __launch_bounds__(256) void cvt_f32_bf16(const float* __restrict__ in,
                                                    bf16* __restrict__ out, int n) {
  int i = blockIdx.x * 256 + threadIdx.x;
  if (i < n) out[i] = (bf16)in[i];
}

// ---------------------------------------------------------------------------
// init: x = E, Asum = 0
// ---------------------------------------------------------------------------
__global__ __launch_bounds__(256) void init_kernel(const float* __restrict__ E,
                                                   float* __restrict__ x,
                                                   float* __restrict__ Asum) {
  size_t i  = (size_t)blockIdx.x * 256 + threadIdx.x;
  const size_t nx = (size_t)M_ * D_;
  const size_t na = (size_t)B_ * T_ * T_;
  if (i < nx) x[i] = E[i];
  if (i < na) Asum[i] = 0.0f;
}

// ---------------------------------------------------------------------------
// WMMA GEMM:  C[M,N] = A[M,K] (bf16) @ W[N,K]^T (bf16) + bias[N]
// Block tile 64(M) x 256(N), 8 waves as 2x4, each wave -> 32x64 (2x4 WMMAs:
// 8 wmma per K-slab per wave vs 6 fragment loads). K staged through LDS in
// 32-wide slabs via async global->LDS copies. Requires M%64==0, N%256==0,
// K%32==0 (true for all four GEMM shapes here).
// ---------------------------------------------------------------------------
template <bool OUT_BF16, bool RELU>
__global__ __launch_bounds__(256) void gemm_wmma(const bf16* __restrict__ A,
                                                 const bf16* __restrict__ W,
                                                 const float* __restrict__ bias,
                                                 float* __restrict__ Cf,
                                                 bf16* __restrict__ Cb,
                                                 int M, int N, int K) {
  __shared__ bf16 As[64][32];    // 4 KB
  __shared__ bf16 Bs[256][32];   // 16 KB

  const int tid  = threadIdx.x;
  const int wave = tid >> 5;
  const int lane = tid & 31;
  const int wm   = wave >> 2;    // 0..1
  const int wn   = wave & 3;     // 0..3
  const int row0 = blockIdx.y * 64;
  const int col0 = blockIdx.x * 256;

  v8f acc[2][4] = {};

  for (int kk = 0; kk < K; kk += 32) {
    // stage A tile: 64 rows x 32 cols, one async 16B chunk per thread
    {
      int r = tid >> 2, c = (tid & 3) * 8;
      async_cp16(&A[(size_t)(row0 + r) * K + kk + c], &As[r][c]);
    }
    // stage W tile: 256 rows x 32 cols, four async 16B chunks per thread
#pragma unroll
    for (int i = 0; i < 4; ++i) {
      int idx = tid + i * 256;
      int r = idx >> 2, c = (idx & 3) * 8;
      async_cp16(&W[(size_t)(col0 + r) * K + kk + c], &Bs[r][c]);
    }
    if (kk + 32 < K) {  // gfx1250 global_prefetch_b8 for the next K slab
      __builtin_prefetch(&A[(size_t)(row0 + (tid >> 2)) * K + kk + 32], 0, 1);
      __builtin_prefetch(&W[(size_t)(col0 + (tid >> 2)) * K + kk + 32], 0, 1);
    }
    wait_async0();
    __syncthreads();

    v16bf af[2], bfr[4];
    const int fr = lane & 15;
#pragma unroll
    for (int i = 0; i < 2; ++i)
      af[i] = make_frag(&As[wm * 32 + i * 16 + fr][0], lane);
#pragma unroll
    for (int j = 0; j < 4; ++j)
      bfr[j] = make_frag(&Bs[wn * 64 + j * 16 + fr][0], lane);
#pragma unroll
    for (int i = 0; i < 2; ++i)
#pragma unroll
      for (int j = 0; j < 4; ++j)
        acc[i][j] = __builtin_amdgcn_wmma_f32_16x16x32_bf16(
            false, af[i], false, bfr[j], (short)0, acc[i][j], false, false);
    __syncthreads();
  }

  // epilogue: C/D layout -> vgpr jj holds row jj (lanes 0-15) / jj+8 (16-31)
  const int colb = col0 + wn * 64;
  const int rowb = row0 + wm * 32 + ((lane < 16) ? 0 : 8);
#pragma unroll
  for (int i = 0; i < 2; ++i)
#pragma unroll
    for (int j = 0; j < 4; ++j) {
      const int cc = colb + j * 16 + (lane & 15);
      const float bb = bias[cc];
#pragma unroll
      for (int jj = 0; jj < 8; ++jj) {
        float v = acc[i][j][jj] + bb;
        if (RELU) v = fmaxf(v, 0.0f);
        size_t off = (size_t)(rowb + i * 16 + jj) * N + cc;
        if constexpr (OUT_BF16) Cb[off] = (bf16)v;
        else                    Cf[off] = v;
      }
    }
}

// ---------------------------------------------------------------------------
// Fused attention for one (batch, 16-query block):
//   loop over heads:
//     S(16x1024) = Q Kt / sqrt(HD)   via WMMA (scores kept f32 in LDS)
//     row softmax (full row in LDS, shfl reductions over 16-lane groups)
//     Asum[b, q, :] += P / (H*L)
//     ctx(16x64)  = P V              via WMMA, per-wave partials merged
//                                    with ds_add_f32 atomics in LDS
// qkv layout: (M_, 3*D_) bf16 with q|k|v column blocks; head h uses cols
// h*64 (q), D_+h*64 (k), 2*D_+h*64 (v). ctx written bf16 (M_, D_).
// ---------------------------------------------------------------------------
__global__ __launch_bounds__(256) void attention_kernel(const bf16* __restrict__ qkv,
                                                        bf16* __restrict__ ctx,
                                                        float* __restrict__ Asum) {
  __shared__ float sc[16][1024];   // 64 KB scores/probs (well under 320KB/WGP)
  __shared__ float ctx_s[16][64];  // 4 KB ctx accumulator

  const int b   = blockIdx.y;
  const int qb  = blockIdx.x;          // 0..63
  const int tid = threadIdx.x;
  const int wave = tid >> 5;
  const int lane = tid & 31;
  const int fr   = lane & 15;
  const int fo   = (lane >> 4) * 8;
  const float scale = 0.125f;          // 1/sqrt(64)
  const float invHL = 1.0f / (H_ * L_);
  const size_t qrow0 = (size_t)b * T_ + (size_t)qb * 16;
  const size_t stride = 3 * D_;        // 1536

  for (int h = 0; h < H_; ++h) {
    // zero ctx accumulator (no conflict with score writes)
    for (int i = tid; i < 16 * 64; i += 256) (&ctx_s[0][0])[i] = 0.0f;

    // ---- scores: each wave owns key tiles kt = wave, wave+8, ... ----
    v16bf q0 = make_frag(qkv + (qrow0 + fr) * stride + h * HD_,      lane);
    v16bf q1 = make_frag(qkv + (qrow0 + fr) * stride + h * HD_ + 32, lane);
    for (int kt = wave; kt < 64; kt += 8) {
      const size_t krow = (size_t)b * T_ + (size_t)kt * 16 + fr;
      v16bf k0 = make_frag(qkv + krow * stride + D_ + h * HD_,      lane);
      v16bf k1 = make_frag(qkv + krow * stride + D_ + h * HD_ + 32, lane);
      v8f s = {};
      s = __builtin_amdgcn_wmma_f32_16x16x32_bf16(false, q0, false, k0, (short)0, s, false, false);
      s = __builtin_amdgcn_wmma_f32_16x16x32_bf16(false, q1, false, k1, (short)0, s, false, false);
      const int rbase = (lane < 16) ? 0 : 8;
#pragma unroll
      for (int jj = 0; jj < 8; ++jj)
        sc[rbase + jj][kt * 16 + fr] = s[jj] * scale;
    }
    __syncthreads();

    // ---- softmax: 16 threads per row, 64 elems each ----
    {
      const int row = tid >> 4;        // 0..15
      const int g   = tid & 15;
      float m = -3.0e38f;
      for (int t = 0; t < 64; ++t) m = fmaxf(m, sc[row][t * 16 + g]);
#pragma unroll
      for (int off = 8; off; off >>= 1) m = fmaxf(m, __shfl_xor(m, off, 32));
      float sum = 0.0f;
      for (int t = 0; t < 64; ++t) {
        float p = __expf(sc[row][t * 16 + g] - m);
        sc[row][t * 16 + g] = p;
        sum += p;
      }
#pragma unroll
      for (int off = 8; off; off >>= 1) sum += __shfl_xor(sum, off, 32);
      const float inv = 1.0f / sum;
      float* arow = Asum + ((size_t)b * T_ + (size_t)qb * 16 + row) * T_;
      for (int t = 0; t < 64; ++t) {
        int c = t * 16 + g;
        float p = sc[row][c] * inv;
        sc[row][c] = p;
        arow[c] += p * invHL;          // block exclusively owns these rows
      }
    }
    __syncthreads();

    // ---- ctx = P @ V : waves split the 32 K-chunks of 32 ----
    {
      v8f vacc[4] = {};
      for (int kc = wave; kc < 32; kc += 8) {
        const int kbase = kc * 32;
        // P fragment (f32 LDS -> bf16)
        V16 pu;
#pragma unroll
        for (int e = 0; e < 8; ++e) {
          pu.v[e]     = (bf16)sc[fr][kbase + fo + e];
          pu.v[8 + e] = (bf16)sc[fr][kbase + 16 + fo + e];
        }
#pragma unroll
        for (int nt = 0; nt < 4; ++nt) {
          // B frag: column n of V == strided walk down V rows
          const bf16* vcol = qkv + ((size_t)b * T_ + kbase) * stride
                           + 2 * D_ + h * HD_ + nt * 16 + fr;
          V16 vu;
#pragma unroll
          for (int e = 0; e < 8; ++e) {
            vu.v[e]     = vcol[(size_t)(fo + e) * stride];
            vu.v[8 + e] = vcol[(size_t)(16 + fo + e) * stride];
          }
          vacc[nt] = __builtin_amdgcn_wmma_f32_16x16x32_bf16(
              false, pu.v, false, vu.v, (short)0, vacc[nt], false, false);
        }
      }
      const int rbase = (lane < 16) ? 0 : 8;
#pragma unroll
      for (int nt = 0; nt < 4; ++nt)
#pragma unroll
        for (int jj = 0; jj < 8; ++jj)
          atomicAdd(&ctx_s[rbase + jj][nt * 16 + fr], vacc[nt][jj]);  // ds_add_f32
    }
    __syncthreads();

    // ---- write ctx tile (16 x 64) as bf16 ----
    for (int i = tid; i < 16 * 64; i += 256) {
      int r = i >> 6, c = i & 63;
      ctx[(qrow0 + r) * D_ + h * HD_ + c] = (bf16)ctx_s[r][c];
    }
    __syncthreads();
  }
}

// ---------------------------------------------------------------------------
// x = LayerNorm(x + y) * g + be   (one 512-elem row per block, 256 threads)
// ---------------------------------------------------------------------------
__global__ __launch_bounds__(256) void add_ln_kernel(float* __restrict__ x,
                                                     const float* __restrict__ y,
                                                     const float* __restrict__ g,
                                                     const float* __restrict__ be) {
  __shared__ float red[16];
  __shared__ float mv[2];
  const int row = blockIdx.x;
  const int tid = threadIdx.x;
  float* xr = x + (size_t)row * D_;
  const float* yr = y + (size_t)row * D_;

  float a = xr[tid]       + yr[tid];
  float c = xr[tid + 256] + yr[tid + 256];
  float s = a + c;
  float q = a * a + c * c;
#pragma unroll
  for (int off = 16; off; off >>= 1) {
    s += __shfl_xor(s, off, 32);
    q += __shfl_xor(q, off, 32);
  }
  const int wave = tid >> 5, lane = tid & 31;
  if (lane == 0) { red[wave] = s; red[8 + wave] = q; }
  __syncthreads();
  if (tid == 0) {
    float S = 0.0f, Q = 0.0f;
#pragma unroll
    for (int w = 0; w < 8; ++w) { S += red[w]; Q += red[8 + w]; }
    float mean = S * (1.0f / D_);
    float var  = Q * (1.0f / D_) - mean * mean;
    mv[0] = mean;
    mv[1] = rsqrtf(var + 1e-5f);
  }
  __syncthreads();
  const float mean = mv[0], inv = mv[1];
  xr[tid]       = (a - mean) * inv * g[tid]       + be[tid];
  xr[tid + 256] = (c - mean) * inv * g[tid + 256] + be[tid + 256];
}

// ---------------------------------------------------------------------------
// host driver
// ---------------------------------------------------------------------------
extern "C" void kernel_launch(void* const* d_in, const int* in_sizes, int n_in,
                              void* d_out, int out_size, void* d_ws, size_t ws_size,
                              hipStream_t stream) {
  const float* E    = (const float*)d_in[0];
  const float* Wqkv = (const float*)d_in[1];
  const float* bqkv = (const float*)d_in[2];
  const float* Wo   = (const float*)d_in[3];
  const float* bo   = (const float*)d_in[4];
  const float* W1   = (const float*)d_in[5];
  const float* b1   = (const float*)d_in[6];
  const float* W2   = (const float*)d_in[7];
  const float* b2   = (const float*)d_in[8];
  const float* g1   = (const float*)d_in[9];
  const float* be1  = (const float*)d_in[10];
  const float* g2   = (const float*)d_in[11];
  const float* be2  = (const float*)d_in[12];

  float* x    = (float*)d_out;                 // (B,T,D) running activations
  float* Asum = x + (size_t)M_ * D_;           // (B,T,T) attention accumulator

  // workspace carve-out (~94 MB)
  char* p = (char*)d_ws;
  bf16*  xbf   = (bf16*)p;  p += (size_t)M_ * D_  * 2;   // 8  MB
  bf16*  wbf   = (bf16*)p;  p += (size_t)FF_ * D_ * 2;   // 2  MB (max weight)
  bf16*  qkvbf = (bf16*)p;  p += (size_t)M_ * 3 * D_ * 2;// 24 MB
  bf16*  ctxbf = (bf16*)p;  p += (size_t)M_ * D_  * 2;   // 8  MB
  bf16*  ffbf  = (bf16*)p;  p += (size_t)M_ * FF_ * 2;   // 32 MB
  float* tmp   = (float*)p;                              // 16 MB

  const dim3 blk(256);
  auto cvt_blocks = [](int n) { return dim3((n + 255) / 256); };

  init_kernel<<<dim3(((size_t)B_ * T_ * T_ + 255) / 256), blk, 0, stream>>>(E, x, Asum);

  for (int l = 0; l < L_; ++l) {
    // ---- fused QKV projection ----
    cvt_f32_bf16<<<cvt_blocks(M_ * D_), blk, 0, stream>>>(x, xbf, M_ * D_);
    cvt_f32_bf16<<<cvt_blocks(3 * D_ * D_), blk, 0, stream>>>(
        Wqkv + (size_t)l * 3 * D_ * D_, wbf, 3 * D_ * D_);
    gemm_wmma<true, false><<<dim3(3 * D_ / 256, M_ / 64), blk, 0, stream>>>(
        xbf, wbf, bqkv + (size_t)l * 3 * D_, nullptr, qkvbf, M_, 3 * D_, D_);

    // ---- fused attention (scores, softmax, Asum, P@V) ----
    attention_kernel<<<dim3(T_ / 16, B_), blk, 0, stream>>>(qkvbf, ctxbf, Asum);

    // ---- output projection + residual LN ----
    cvt_f32_bf16<<<cvt_blocks(D_ * D_), blk, 0, stream>>>(
        Wo + (size_t)l * D_ * D_, wbf, D_ * D_);
    gemm_wmma<false, false><<<dim3(D_ / 256, M_ / 64), blk, 0, stream>>>(
        ctxbf, wbf, bo + (size_t)l * D_, tmp, nullptr, M_, D_, D_);
    add_ln_kernel<<<dim3(M_), blk, 0, stream>>>(x, tmp, g1 + (size_t)l * D_, be1 + (size_t)l * D_);

    // ---- FFN ----
    cvt_f32_bf16<<<cvt_blocks(M_ * D_), blk, 0, stream>>>(x, xbf, M_ * D_);
    cvt_f32_bf16<<<cvt_blocks(FF_ * D_), blk, 0, stream>>>(
        W1 + (size_t)l * FF_ * D_, wbf, FF_ * D_);
    gemm_wmma<true, true><<<dim3(FF_ / 256, M_ / 64), blk, 0, stream>>>(
        xbf, wbf, b1 + (size_t)l * FF_, nullptr, ffbf, M_, FF_, D_);
    cvt_f32_bf16<<<cvt_blocks(D_ * FF_), blk, 0, stream>>>(
        W2 + (size_t)l * D_ * FF_, wbf, D_ * FF_);
    gemm_wmma<false, false><<<dim3(D_ / 256, M_ / 64), blk, 0, stream>>>(
        ffbf, wbf, b2 + (size_t)l * D_, tmp, nullptr, M_, D_, FF_);
    add_ln_kernel<<<dim3(M_), blk, 0, stream>>>(x, tmp, g2 + (size_t)l * D_, be2 + (size_t)l * D_);
  }
}
